// MHSA_5970004541819
// MI455X (gfx1250) — matrix operand order repaired and checked by reading, not compiled, not run on hardware
//
#include <hip/hip_runtime.h>

typedef __attribute__((ext_vector_type(16))) _Float16 v16h;
typedef __attribute__((ext_vector_type(8)))  float    v8f;
typedef __attribute__((ext_vector_type(4)))  unsigned int u32x4;
typedef __attribute__((ext_vector_type(8)))  unsigned int u32x8;

#define WMMA16(a, b, c) \
  __builtin_amdgcn_wmma_f32_16x16x32_f16(false, (a), false, (b), (short)0, (c), false, false)

#define BGN   128     // B*G
#define TT    256     // T
#define CC    256     // C
#define HH    4
#define HD    64
#define OUTC  512
#define MROWS 32768   // BGN*TT

// ---- Tensor Data Mover: 2D f16 tile (rows x cols), row-major source with
// element stride row_stride_elems, into contiguous LDS at lds_off. One wave
// issues; tracked by TENSORcnt. D# layout per CDNA5 ISA.
__device__ __forceinline__ void tdm_load_tile_f16(
    unsigned lds_off, const _Float16* gsrc,
    unsigned rows, unsigned cols, unsigned row_stride_elems) {
  unsigned long long ga = (unsigned long long)(uintptr_t)gsrc;
  u32x4 g0;
  g0[0] = 1u;                                           // count=1, user D#
  g0[1] = lds_off;                                      // lds_addr
  g0[2] = (unsigned)ga;                                 // global_addr[31:0]
  g0[3] = ((unsigned)(ga >> 32) & 0x01FFFFFFu)          // global_addr[56:32]
          | 0x80000000u;                                // type=2 ("image")
  u32x8 g1;
  g1[0] = 1u << 16;                                     // data_size=2B; mask=0
  g1[1] = (cols & 0xFFFFu) << 16;                       // tensor_dim0[15:0]
  g1[2] = ((cols >> 16) & 0xFFFFu)                      // tensor_dim0[31:16]
          | ((rows & 0xFFFFu) << 16);                   // tensor_dim1[15:0]
  g1[3] = ((rows >> 16) & 0xFFFFu)                      // tensor_dim1[31:16]
          | ((cols & 0xFFFFu) << 16);                   // tile_dim0
  g1[4] = rows & 0xFFFFu;                               // tile_dim1; tile_dim2=0
  g1[5] = row_stride_elems;                             // tensor_dim0_stride lo
  g1[6] = 0u;                                           // stride hi | dim1_stride lo
  g1[7] = 0u;
  asm volatile("tensor_load_to_lds %0, %1" :: "s"(g0), "s"(g1) : "memory");
}

// ---- fragment loaders (CDNA5 wave32 WMMA 16x16x32 f16; contiguous k order, same for A and B) ----
// A (16x32, MxK): lane L: row M = L&15, half p = L>>4; halfword j -> K = j + 16p
__device__ __forceinline__ v16h afrag_lds(const _Float16* base, int ldk, int m, int p) {
  v16h a;
  const _Float16* row = base + (size_t)m * ldk;
#pragma unroll
  for (int j = 0; j < 16; ++j) a[j] = row[16 * p + j];
  return a;
}
// B (32x16, KxN): lane L: col N = L&15, half p = L>>4; halfword j -> K = j + 16p
__device__ __forceinline__ v16h bfrag_lds(const _Float16* base, int ldk, int n, int p) {
  v16h b;
  const _Float16* row = base + (size_t)n * ldk + 16 * p;
#pragma unroll
  for (int j = 0; j < 16; ++j) b[j] = row[j];
  return b;
}

// ---------------- Kernel 1: fused QKV GEMM ----------------
// q/k/v = x @ W^T + b : M=32768, N=768 (q|k|v), K=256. f32 source -> convert in LDS.
__global__ void __launch_bounds__(256) qkv_gemm(
    const float* __restrict__ X,
    const float* __restrict__ Wq, const float* __restrict__ bq,
    const float* __restrict__ Wk, const float* __restrict__ bk,
    const float* __restrict__ Wv, const float* __restrict__ bv,
    _Float16* __restrict__ qh, _Float16* __restrict__ kh, _Float16* __restrict__ vh) {
  __shared__ _Float16 As[128][32];
  __shared__ _Float16 Bs[64][32];
  const int tid = threadIdx.x, wave = tid >> 5, lane = tid & 31;
  const int p = lane >> 4, ln = lane & 15;
  const int m0 = blockIdx.x * 128;
  const int gn0 = blockIdx.y * 64;
  const int mat = gn0 >> 8;
  const int n0 = gn0 & 255;
  const float* W    = (mat == 0) ? Wq : ((mat == 1) ? Wk : Wv);
  const float* bias = (mat == 0) ? bq : ((mat == 1) ? bk : bv);
  _Float16* outp    = (mat == 0) ? qh : ((mat == 1) ? kh : vh);

  v8f acc[4] = {};
  for (int k0 = 0; k0 < CC; k0 += 32) {
    for (int i = tid; i < 128 * 32; i += 256) {
      int r = i >> 5, c = i & 31;
      As[r][c] = (_Float16)X[(size_t)(m0 + r) * CC + k0 + c];
    }
    for (int i = tid; i < 64 * 32; i += 256) {
      int r = i >> 5, c = i & 31;
      Bs[r][c] = (_Float16)W[(size_t)(n0 + r) * CC + k0 + c];
    }
    __syncthreads();
    v16h a = afrag_lds(&As[wave * 16][0], 32, ln, p);
#pragma unroll
    for (int nt = 0; nt < 4; ++nt) {
      v16h b = bfrag_lds(&Bs[nt * 16][0], 32, ln, p);
      acc[nt] = WMMA16(a, b, acc[nt]);
    }
    __syncthreads();
  }
#pragma unroll
  for (int nt = 0; nt < 4; ++nt) {
    int col = n0 + nt * 16 + ln;
    float bb = bias[col];
    size_t row0 = (size_t)m0 + wave * 16 + 8 * p;
#pragma unroll
    for (int r = 0; r < 8; ++r)
      outp[(row0 + r) * CC + col] = (_Float16)(acc[nt][r] + bb);
  }
}

// ---------------- Kernel 2: flash attention ----------------
// per (bg, head, 64-row block). 4 waves, wave = 16 rows. K/V tiles staged by TDM.
__global__ void __launch_bounds__(128) attn_kernel(
    const _Float16* __restrict__ qh, const _Float16* __restrict__ kh,
    const _Float16* __restrict__ vh, _Float16* __restrict__ avh) {
  __shared__ _Float16 Kb[32][64];
  __shared__ _Float16 Vb[32][64];
  __shared__ _Float16 Pst[4][16][32];
  const int tid = threadIdx.x, wave = tid >> 5, lane = tid & 31;
  const int p = lane >> 4, ln = lane & 15;
  const int bg = blockIdx.x, h = blockIdx.y, mb = blockIdx.z;
  const int t0 = mb * 64 + wave * 16;

  // Q fragments (16 rows x 64 d) straight from global f16
  const _Float16* qrow = qh + ((size_t)bg * TT + t0 + ln) * CC + h * HD;
  v16h aQ[2];
#pragma unroll
  for (int kf = 0; kf < 2; ++kf)
#pragma unroll
    for (int j = 0; j < 16; ++j) {
      int d = kf * 32 + 16 * p + j;
      aQ[kf][j] = qrow[d];
    }

  v8f o[4] = {};
  float mrow[8], lrow[8];
#pragma unroll
  for (int r = 0; r < 8; ++r) { mrow[r] = -1e30f; lrow[r] = 0.0f; }

  for (int sb = 0; sb < 8; ++sb) {
    const int s0 = sb * 32;
    if (wave == 0) {
      size_t gbase = ((size_t)bg * TT + s0) * CC + h * HD;
      tdm_load_tile_f16((unsigned)(uintptr_t)&Kb[0][0], kh + gbase, 32, 64, CC);
      tdm_load_tile_f16((unsigned)(uintptr_t)&Vb[0][0], vh + gbase, 32, 64, CC);
      __builtin_amdgcn_s_wait_tensorcnt(0);
    }
    __syncthreads();

    // S = Q K^T / 8 : 16 rows x 32 cols per wave (2 n-tiles)
    v8f s[2] = {};
#pragma unroll
    for (int kf = 0; kf < 2; ++kf) {
#pragma unroll
      for (int nt = 0; nt < 2; ++nt) {
        v16h bK;
#pragma unroll
        for (int j = 0; j < 16; ++j) bK[j] = Kb[nt * 16 + ln][kf * 32 + 16 * p + j];
        s[nt] = WMMA16(aQ[kf], bK, s[nt]);
      }
    }

    float mx[8], rs[8], alpha[8];
#pragma unroll
    for (int r = 0; r < 8; ++r) {
      s[0][r] *= 0.125f; s[1][r] *= 0.125f;
      float v = fmaxf(s[0][r], s[1][r]);
      v = fmaxf(v, __shfl_xor(v, 1, 32));
      v = fmaxf(v, __shfl_xor(v, 2, 32));
      v = fmaxf(v, __shfl_xor(v, 4, 32));
      v = fmaxf(v, __shfl_xor(v, 8, 32));
      mx[r] = v;
    }
#pragma unroll
    for (int r = 0; r < 8; ++r) {
      float mn = fmaxf(mrow[r], mx[r]);
      alpha[r] = __expf(mrow[r] - mn);
      mrow[r] = mn;
      float e0 = __expf(s[0][r] - mn);
      float e1 = __expf(s[1][r] - mn);
      s[0][r] = e0; s[1][r] = e1;
      float v = e0 + e1;
      v += __shfl_xor(v, 1, 32);
      v += __shfl_xor(v, 2, 32);
      v += __shfl_xor(v, 4, 32);
      v += __shfl_xor(v, 8, 32);
      rs[r] = v;
      lrow[r] = lrow[r] * alpha[r] + rs[r];
    }
#pragma unroll
    for (int dt = 0; dt < 4; ++dt)
#pragma unroll
      for (int r = 0; r < 8; ++r) o[dt][r] *= alpha[r];

    // restage P (C-layout -> A-layout) through LDS
#pragma unroll
    for (int nt = 0; nt < 2; ++nt)
#pragma unroll
      for (int r = 0; r < 8; ++r)
        Pst[wave][r + 8 * p][nt * 16 + ln] = (_Float16)s[nt][r];
    __syncthreads();

    v16h aP = afrag_lds(&Pst[wave][0][0], 32, ln, p);
#pragma unroll
    for (int dt = 0; dt < 4; ++dt) {
      v16h bV;
#pragma unroll
      for (int j = 0; j < 16; ++j) bV[j] = Vb[16 * p + j][dt * 16 + ln];
      o[dt] = WMMA16(aP, bV, o[dt]);
    }
    __syncthreads();
  }

#pragma unroll
  for (int dt = 0; dt < 4; ++dt)
#pragma unroll
    for (int r = 0; r < 8; ++r) {
      size_t g = ((size_t)bg * TT + t0 + r + 8 * p) * CC + h * HD + dt * 16 + ln;
      avh[g] = (_Float16)(o[dt][r] / lrow[r]);
    }
}

// ---------------- Kernel 3: value GEMM ----------------
// value = av @ Wo^T + bo : M=32768, N=512, K=256. A tile via TDM (f16 source).
__global__ void __launch_bounds__(256) value_gemm(
    const _Float16* __restrict__ A, const float* __restrict__ Wo,
    const float* __restrict__ bo, float* __restrict__ value,
    _Float16* __restrict__ valueh) {
  __shared__ _Float16 As[128][32];
  __shared__ _Float16 Bs[64][32];
  const int tid = threadIdx.x, wave = tid >> 5, lane = tid & 31;
  const int p = lane >> 4, ln = lane & 15;
  const int m0 = blockIdx.x * 128;
  const int n0 = blockIdx.y * 64;
  v8f acc[4] = {};
  for (int k0 = 0; k0 < CC; k0 += 32) {
    if (wave == 0)
      tdm_load_tile_f16((unsigned)(uintptr_t)&As[0][0],
                        A + (size_t)m0 * CC + k0, 128, 32, CC);
    for (int i = tid; i < 64 * 32; i += 256) {
      int r = i >> 5, c = i & 31;
      Bs[r][c] = (_Float16)Wo[(size_t)(n0 + r) * CC + k0 + c];
    }
    if (wave == 0) __builtin_amdgcn_s_wait_tensorcnt(0);
    __syncthreads();
    v16h a = afrag_lds(&As[wave * 16][0], 32, ln, p);
#pragma unroll
    for (int nt = 0; nt < 4; ++nt) {
      v16h b = bfrag_lds(&Bs[nt * 16][0], 32, ln, p);
      acc[nt] = WMMA16(a, b, acc[nt]);
    }
    __syncthreads();
  }
#pragma unroll
  for (int nt = 0; nt < 4; ++nt) {
    int col = n0 + nt * 16 + ln;
    float bb = bo[col];
    size_t row0 = (size_t)m0 + wave * 16 + 8 * p;
#pragma unroll
    for (int r = 0; r < 8; ++r) {
      float v = acc[nt][r] + bb;
      value[(row0 + r) * OUTC + col] = v;
      valueh[(row0 + r) * OUTC + col] = (_Float16)v;
    }
  }
}

// ---------------- Kernel 4: co -> zc reduction ----------------
// zc[s] = mean_c( y[c] + value[s,:] . Woff[c,:] + boff[c] ), c in 0..2
__global__ void __launch_bounds__(256) cozc_kernel(
    const float* __restrict__ value, const float* __restrict__ Woff,
    const float* __restrict__ boff, const float* __restrict__ y,
    float* __restrict__ zc) {
  const int tid = threadIdx.x, wave = tid >> 5, lane = tid & 31;
  const size_t row = (size_t)blockIdx.x * 8 + wave;
  const float* vr = value + row * OUTC;
  float s0 = 0.f, s1 = 0.f, s2 = 0.f;
  for (int i = lane; i < OUTC; i += 32) {
    float v = vr[i];
    s0 += v * Woff[i];
    s1 += v * Woff[OUTC + i];
    s2 += v * Woff[2 * OUTC + i];
  }
#pragma unroll
  for (int m = 1; m < 32; m <<= 1) {
    s0 += __shfl_xor(s0, m, 32);
    s1 += __shfl_xor(s1, m, 32);
    s2 += __shfl_xor(s2, m, 32);
  }
  if (lane == 0) {
    float c0 = s0 + boff[0], c1 = s1 + boff[1], c2 = s2 + boff[2];
    zc[row] = ((y[0] + c0) + (y[1] + c1) + (y[2] + c2)) * (1.0f / 3.0f);
  }
}

// ---------------- Kernel 5: off GEMM + BN + add value ----------------
// z = value + ((value @ Woff[3:].T + boff[3:]) - rmean)*rsqrt(rvar+eps)*gamma + beta
__global__ void __launch_bounds__(256) off_gemm(
    const _Float16* __restrict__ A, const float* __restrict__ Woff,
    const float* __restrict__ boff, const float* __restrict__ gamma,
    const float* __restrict__ beta, const float* __restrict__ rmean,
    const float* __restrict__ rvar, const float* __restrict__ value,
    float* __restrict__ z) {
  __shared__ _Float16 As[128][32];
  __shared__ _Float16 Bs[64][32];
  const int tid = threadIdx.x, wave = tid >> 5, lane = tid & 31;
  const int p = lane >> 4, ln = lane & 15;
  const int m0 = blockIdx.x * 128;
  const int n0 = blockIdx.y * 64;
  v8f acc[4] = {};
  for (int k0 = 0; k0 < OUTC; k0 += 32) {
    if (wave == 0)
      tdm_load_tile_f16((unsigned)(uintptr_t)&As[0][0],
                        A + (size_t)m0 * OUTC + k0, 128, 32, OUTC);
    for (int i = tid; i < 64 * 32; i += 256) {
      int r = i >> 5, c = i & 31;
      Bs[r][c] = (_Float16)Woff[(size_t)(3 + n0 + r) * OUTC + k0 + c];
    }
    if (wave == 0) __builtin_amdgcn_s_wait_tensorcnt(0);
    __syncthreads();
    v16h a = afrag_lds(&As[wave * 16][0], 32, ln, p);
#pragma unroll
    for (int nt = 0; nt < 4; ++nt) {
      v16h b = bfrag_lds(&Bs[nt * 16][0], 32, ln, p);
      acc[nt] = WMMA16(a, b, acc[nt]);
    }
    __syncthreads();
  }
#pragma unroll
  for (int nt = 0; nt < 4; ++nt) {
    int col = n0 + nt * 16 + ln;
    float bb = boff[3 + col];
    float sc = rsqrtf(rvar[col] + 1e-5f) * gamma[col];
    float rm = rmean[col], bt = beta[col];
    size_t row0 = (size_t)m0 + wave * 16 + 8 * p;
#pragma unroll
    for (int r = 0; r < 8; ++r) {
      float off = acc[nt][r] + bb;
      float fo = (off - rm) * sc + bt;
      size_t idx = (row0 + r) * OUTC + col;
      z[idx] = value[idx] + fo;
    }
  }
}

// ---------------- Kernel 6: suffix-sum finalize ----------------
// mask[t][s] = (t>=88) && (s>t); denom[t]=255-t; out[...,0:512]=suffsum(z)/denom,
// out[...,512] = suffsum(zc)/denom * 10
__global__ void __launch_bounds__(256) finalize_kernel(
    const float* __restrict__ z, const float* __restrict__ zc,
    float* __restrict__ out) {
  const int tid = threadIdx.x;
  const int bg = blockIdx.x;
  const float* zb = z + (size_t)bg * TT * OUTC;
  float* ob = out + (size_t)bg * TT * 513;
  float a0 = 0.f, a1 = 0.f, ac = 0.f;
  for (int s = TT - 1; s >= 1; --s) {
    const float* zr = zb + (size_t)s * OUTC;
    a0 += zr[tid];
    a1 += zr[tid + 256];
    ac += zc[(size_t)bg * TT + s];
    int t = s - 1;
    if (t >= 88 && t <= 254) {
      float inv = 1.0f / (float)(255 - t);
      float* orow = ob + (size_t)t * 513;
      orow[tid] = a0 * inv;
      orow[tid + 256] = a1 * inv;
      if (tid == 0) orow[512] = ac * inv * 10.0f;
    }
  }
  for (int t = 0; t < 88; ++t) {
    float* orow = ob + (size_t)t * 513;
    orow[tid] = 0.f; orow[tid + 256] = 0.f;
    if (tid == 0) orow[512] = 0.f;
  }
  {
    float* orow = ob + (size_t)255 * 513;
    orow[tid] = 0.f; orow[tid + 256] = 0.f;
    if (tid == 0) orow[512] = 0.f;
  }
}

extern "C" void kernel_launch(void* const* d_in, const int* in_sizes, int n_in,
                              void* d_out, int out_size, void* d_ws, size_t ws_size,
                              hipStream_t stream) {
  (void)in_sizes; (void)n_in; (void)out_size; (void)ws_size;
  const float* x     = (const float*)d_in[0];
  const float* y     = (const float*)d_in[1];
  const float* Wq    = (const float*)d_in[2];
  const float* bq    = (const float*)d_in[3];
  const float* Wk    = (const float*)d_in[4];
  const float* bk    = (const float*)d_in[5];
  const float* Wv    = (const float*)d_in[6];
  const float* bv    = (const float*)d_in[7];
  const float* Wo    = (const float*)d_in[8];
  const float* bo    = (const float*)d_in[9];
  const float* Woff  = (const float*)d_in[10];
  const float* boff  = (const float*)d_in[11];
  const float* gamma = (const float*)d_in[12];
  const float* beta  = (const float*)d_in[13];
  const float* rmean = (const float*)d_in[14];
  const float* rvar  = (const float*)d_in[15];
  float* out = (float*)d_out;

  char* ws = (char*)d_ws;
  size_t off = 0;
  _Float16* qh     = (_Float16*)(ws + off); off += (size_t)MROWS * CC * 2;
  _Float16* kh     = (_Float16*)(ws + off); off += (size_t)MROWS * CC * 2;
  _Float16* vh     = (_Float16*)(ws + off); off += (size_t)MROWS * CC * 2;
  _Float16* avh    = (_Float16*)(ws + off); off += (size_t)MROWS * CC * 2;
  _Float16* valueh = (_Float16*)(ws + off); off += (size_t)MROWS * OUTC * 2;
  float*    value  = (float*)(ws + off);    off += (size_t)MROWS * OUTC * 4;
  float*    z      = (float*)(ws + off);    off += (size_t)MROWS * OUTC * 4;
  float*    zc     = (float*)(ws + off);    off += (size_t)MROWS * 4;

  dim3 g1(MROWS / 128, 12);
  qkv_gemm<<<g1, 256, 0, stream>>>(x, Wq, bq, Wk, bk, Wv, bv, qh, kh, vh);

  dim3 g2(BGN, HH, 4);
  attn_kernel<<<g2, 128, 0, stream>>>(qh, kh, vh, avh);

  dim3 g3(MROWS / 128, OUTC / 64);
  value_gemm<<<g3, 256, 0, stream>>>(avh, Wo, bo, value, valueh);

  cozc_kernel<<<MROWS / 8, 256, 0, stream>>>(value, Woff, boff, y, zc);

  dim3 g5(MROWS / 128, OUTC / 64);
  off_gemm<<<g5, 256, 0, stream>>>(valueh, Woff, boff, gamma, beta, rmean, rvar,
                                   value, z);

  finalize_kernel<<<BGN, 256, 0, stream>>>(z, zc, out);
}
